// EnsembleSharedVQC_52613349376577
// MI455X (gfx1250) — matrix operand order repaired and checked by reading, not compiled, not run on hardware
//
#include <hip/hip_runtime.h>
#include <math.h>

// ---------------------------------------------------------------------------
// EnsembleSharedVQC on MI455X (gfx1250)
//   V (4096x4096 complex) = shared variational unitary (built per basis col)
//   S (4096x4096 complex) = product-state encodings (4 encoders x 1024 batch)
//   Y = V * S  as complex GEMM in split-bf16 (hi/lo error compensation):
//     each real GEMM = Ah*Bh + Ah*Bl + Al*Bh  via V_WMMA_F32_16X16X32_BF16
//   -> fp32-class accuracy at the bf16 matrix-pipe rate; bf16 planes put
//      V+S into 256MB (8 planes x 32MB).
//   fused |Y|^2 -> per-qubit signed reduction -> softmax(alpha) combine
// ---------------------------------------------------------------------------

typedef __attribute__((ext_vector_type(16))) __bf16 v16bf;
typedef __attribute__((ext_vector_type(8)))  float  v8f;

#define NQ    12
#define DIM   4096          // 2^NQ
#define NB    1024
#define NE    4
#define NL    4
#define NCIRC (NE * NB)     // 4096 circuits

// ------------------------- bf16 split helpers ------------------------------

__device__ __forceinline__ unsigned short f32_to_bf16(float x) {
  union { float f; unsigned int u; } v; v.f = x;
  unsigned int u = v.u;
  return (unsigned short)((u + 0x7FFFu + ((u >> 16) & 1u)) >> 16);   // RTNE
}
__device__ __forceinline__ float bf16_to_f32(unsigned short h) {
  union { unsigned int u; float f; } v; v.u = ((unsigned int)h) << 16;
  return v.f;
}
__device__ __forceinline__ void split_bf16(float x, unsigned short& hi, unsigned short& lo) {
  hi = f32_to_bf16(x);
  lo = f32_to_bf16(x - bf16_to_f32(hi));
}
__device__ __forceinline__ unsigned int pack2(unsigned short a, unsigned short b) {
  return (unsigned int)a | ((unsigned int)b << 16);
}

// ------------------------- gate helpers (LDS state) ------------------------

__device__ __forceinline__ void pair_idx(int k, int p, int& i0, int& i1) {
  int mask = (1 << p) - 1;
  i0 = ((k >> p) << (p + 1)) | (k & mask);
  i1 = i0 | (1 << p);
}

__device__ __forceinline__ void gate_rx(float* sre, float* sim, int p, float c, float s, int tid) {
  __syncthreads();
  for (int k = tid; k < DIM / 2; k += 256) {
    int i0, i1; pair_idx(k, p, i0, i1);
    float ar = sre[i0], ai = sim[i0], br = sre[i1], bi = sim[i1];
    sre[i0] = c * ar + s * bi;  sim[i0] = c * ai - s * br;
    sre[i1] = c * br + s * ai;  sim[i1] = c * bi - s * ar;
  }
}

__device__ __forceinline__ void gate_ry(float* sre, float* sim, int p, float c, float s, int tid) {
  __syncthreads();
  for (int k = tid; k < DIM / 2; k += 256) {
    int i0, i1; pair_idx(k, p, i0, i1);
    float ar = sre[i0], ai = sim[i0], br = sre[i1], bi = sim[i1];
    sre[i0] = c * ar - s * br;  sim[i0] = c * ai - s * bi;
    sre[i1] = s * ar + c * br;  sim[i1] = s * ai + c * bi;
  }
}

__device__ __forceinline__ void gate_rz(float* sre, float* sim, int p, float c, float s, int tid) {
  __syncthreads();
  for (int i = tid; i < DIM; i += 256) {
    float xr = sre[i], xi = sim[i];
    float u = ((i >> p) & 1) ? s : -s;   // bit0: e^{-it/2}, bit1: e^{+it/2}
    sre[i] = c * xr - u * xi;
    sim[i] = c * xi + u * xr;
  }
}

__device__ __forceinline__ void gate_cnot(float* sre, float* sim, int q, int tid) {
  const int pt = 10 - q;                  // target bit pos; control = pt+1
  __syncthreads();
  for (int k = tid; k < DIM / 4; k += 256) {
    int low  = k & ((1 << pt) - 1);
    int high = k >> pt;
    int i0 = (high << (pt + 2)) | (1 << (pt + 1)) | low;   // ctrl=1, tgt=0
    int i1 = i0 | (1 << pt);                               // ctrl=1, tgt=1
    float tr = sre[i0], ti = sim[i0];
    sre[i0] = sre[i1]; sim[i0] = sim[i1];
    sre[i1] = tr;      sim[i1] = ti;
  }
}

__device__ __forceinline__ void apply_variational(float* sre, float* sim,
                                                  const float* __restrict__ theta, int tid) {
  for (int l = 0; l < NL; ++l) {
    for (int q = 0; q < NQ; ++q) {
      const int p = 11 - q;
      float t0 = theta[(l * NQ + q) * 3 + 0];
      float t1 = theta[(l * NQ + q) * 3 + 1];
      float t2 = theta[(l * NQ + q) * 3 + 2];
      gate_rx(sre, sim, p, cosf(t0 * 0.5f), sinf(t0 * 0.5f), tid);
      gate_ry(sre, sim, p, cosf(t1 * 0.5f), sinf(t1 * 0.5f), tid);
      gate_rz(sre, sim, p, cosf(t2 * 0.5f), sinf(t2 * 0.5f), tid);
    }
    for (int q = 0; q < NQ - 1; q += 2) gate_cnot(sre, sim, q, tid);
    for (int q = 1; q < NQ - 1; q += 2) gate_cnot(sre, sim, q, tid);
  }
}

__device__ __forceinline__ void enc_vec(int e, float f,
                                        float& v0r, float& v0i, float& v1r, float& v1i) {
  const float r = 0.70710678118654752f;
  float c = cosf(f * 0.5f), s = sinf(f * 0.5f);
  if (e == 0)      { v0r = c;           v0i = 0.f;     v1r = 0.f;         v1i = -s; }
  else if (e == 1) { v0r = c;           v0i = 0.f;     v1r = s;           v1i = 0.f; }
  else if (e == 2) { v0r = c * r;       v0i = -s * r;  v1r = c * r;       v1i = -s * r; }
  else             { v0r = (c - s) * r; v0i = 0.f;     v1r = (c + s) * r; v1i = 0.f; }
}

// ------------------------- kernel A: build V^T (split bf16) ----------------
// block c evolves basis state e_c; stores V[i][c] at plane[c*DIM + i] (k-major)

__global__ __launch_bounds__(256) void build_v_kernel(const float* __restrict__ theta,
                                                      unsigned short* __restrict__ vrh,
                                                      unsigned short* __restrict__ vrl,
                                                      unsigned short* __restrict__ vih,
                                                      unsigned short* __restrict__ vil) {
  __shared__ float sre[DIM];
  __shared__ float sim[DIM];
  const int c = blockIdx.x;
  const int tid = threadIdx.x;
  for (int i = tid; i < DIM; i += 256) { sre[i] = (i == c) ? 1.f : 0.f; sim[i] = 0.f; }
  apply_variational(sre, sim, theta, tid);
  __syncthreads();
  const size_t base = (size_t)c * DIM;
  for (int i = 2 * tid; i < DIM; i += 512) {
    unsigned short h0, l0, h1, l1;
    split_bf16(sre[i],     h0, l0);
    split_bf16(sre[i + 1], h1, l1);
    *(unsigned int*)(vrh + base + i) = pack2(h0, h1);
    *(unsigned int*)(vrl + base + i) = pack2(l0, l1);
    split_bf16(sim[i],     h0, l0);
    split_bf16(sim[i + 1], h1, l1);
    *(unsigned int*)(vih + base + i) = pack2(h0, h1);
    *(unsigned int*)(vil + base + i) = pack2(l0, l1);
  }
}

// ------------------------- kernel B: build S^T (split bf16) ----------------
// block j computes circuit j's product state; stores S[k][j] at plane[j*DIM+k]

__global__ __launch_bounds__(256) void build_s_kernel(const float* __restrict__ features,
                                                      unsigned short* __restrict__ srh,
                                                      unsigned short* __restrict__ srl,
                                                      unsigned short* __restrict__ sih,
                                                      unsigned short* __restrict__ sil) {
  __shared__ float vr[NQ][2], vi[NQ][2];
  const int j = blockIdx.x;
  const int e = j >> 10;
  const int b = j & (NB - 1);
  const int tid = threadIdx.x;
  if (tid < NQ) {
    float v0r, v0i, v1r, v1i;
    enc_vec(e, features[b * NQ + tid], v0r, v0i, v1r, v1i);
    vr[tid][0] = v0r; vi[tid][0] = v0i;
    vr[tid][1] = v1r; vi[tid][1] = v1i;
  }
  __syncthreads();
  const size_t base = (size_t)j * DIM;
  for (int i0 = 2 * tid; i0 < DIM; i0 += 512) {
    float prv[2], piv[2];
#pragma unroll
    for (int t = 0; t < 2; ++t) {
      int i = i0 + t;
      float pr = 1.f, pi = 0.f;
#pragma unroll
      for (int q = 0; q < NQ; ++q) {
        int bit = (i >> (11 - q)) & 1;
        float wr_ = vr[q][bit], wi_ = vi[q][bit];
        float nr = pr * wr_ - pi * wi_;
        float ni = pr * wi_ + pi * wr_;
        pr = nr; pi = ni;
      }
      prv[t] = pr; piv[t] = pi;
    }
    unsigned short h0, l0, h1, l1;
    split_bf16(prv[0], h0, l0); split_bf16(prv[1], h1, l1);
    *(unsigned int*)(srh + base + i0) = pack2(h0, h1);
    *(unsigned int*)(srl + base + i0) = pack2(l0, l1);
    split_bf16(piv[0], h0, l0); split_bf16(piv[1], h1, l1);
    *(unsigned int*)(sih + base + i0) = pack2(h0, h1);
    *(unsigned int*)(sil + base + i0) = pack2(l0, l1);
  }
}

// ------------------------- kernel C: split-bf16 complex GEMM + fused Z -----
// 256 threads = 8 waves; block tile 64x64, K-tile 32; wave tile 16x32.
// Re += Ah*Bh + Ah*Bl + Al*Bh  (re*re) minus same for (im*im) [A-imag negated
// in VALU: packed xor 0x80008000, since bf16 WMMA NEG bits must be 0]
// Im += (Are hi/lo x Bim) + (Aim hi/lo x Bre).   12 WMMAs per subtile per K=32.

#define BM 64
#define BN 64
#define BK 32
#define LDP 40   // row padding (u16): 80B rows -> 16B aligned, bank-spread

union F16x16 {
  uint4 q[2];
  unsigned int u[8];
  v16bf v;
};

__device__ __forceinline__ v8f wmma_bf16(v16bf a, v16bf b, v8f c) {
  return __builtin_amdgcn_wmma_f32_16x16x32_bf16(false, a, false, b, (short)0, c, false, false);
}

__global__ __launch_bounds__(256) void gemm_z_kernel(const unsigned short* __restrict__ vrh,
                                                     const unsigned short* __restrict__ vrl,
                                                     const unsigned short* __restrict__ vih,
                                                     const unsigned short* __restrict__ vil,
                                                     const unsigned short* __restrict__ srh,
                                                     const unsigned short* __restrict__ srl,
                                                     const unsigned short* __restrict__ sih,
                                                     const unsigned short* __restrict__ sil,
                                                     float* __restrict__ Eacc) {
  __shared__ __align__(16) unsigned short Arh[BM][LDP], Arl[BM][LDP];
  __shared__ __align__(16) unsigned short Aih[BM][LDP], Ail[BM][LDP];
  __shared__ __align__(16) unsigned short Brh[BN][LDP], Brl[BN][LDP];
  __shared__ __align__(16) unsigned short Bih[BN][LDP], Bil[BN][LDP];
  __shared__ float ez[BN][NQ];

  const int tid  = threadIdx.x;
  const int wv   = tid >> 5;
  const int lane = tid & 31;
  const int m0 = blockIdx.x * BM;
  const int n0 = blockIdx.y * BN;

  for (int idx = tid; idx < BN * NQ; idx += 256) ((float*)ez)[idx] = 0.f;

  const int ml  = (wv & 3) * 16;       // wave's local M offset
  const int nwl = (wv >> 2) * 32;      // wave's local N offset
  const int lm  = lane & 15;
  const int ah  = (lane >> 4) << 3;    // A frag u16 offset: 0 | 8
  const int bh  = (lane >> 4) << 3;    // B frag u16 offset: 0 | 8

  v8f accr0 = {0.f,0.f,0.f,0.f,0.f,0.f,0.f,0.f};
  v8f acci0 = accr0, accr1 = accr0, acci1 = accr0;

  // A tile fill: global k-major [k][m] -> LDS [m][k] via 2x2 u16 transpose
#define FILL_A(SRC, DST)                                                      \
  for (int idx = tid; idx < (BK / 2) * (BM / 2); idx += 256) {                \
    int kk2 = idx >> 5, mm2 = idx & 31;                                       \
    size_t g = (size_t)(k0 + 2 * kk2) * DIM + (m0 + 2 * mm2);                 \
    unsigned int a0 = *(const unsigned int*)(SRC + g);                        \
    unsigned int a1 = *(const unsigned int*)(SRC + g + DIM);                  \
    *(unsigned int*)(&DST[2 * mm2][2 * kk2])     = (a0 & 0xFFFFu) | (a1 << 16); \
    *(unsigned int*)(&DST[2 * mm2 + 1][2 * kk2]) = (a0 >> 16) | (a1 & 0xFFFF0000u); \
  }
  // B tile fill: global [n][k] (k contiguous) -> LDS [n][k], straight uint4
#define FILL_B(SRC, DST)                                                      \
  for (int idx = tid; idx < BN * (BK / 8); idx += 256) {                      \
    int nn = idx >> 2, kq = (idx & 3) * 8;                                    \
    *(uint4*)(&DST[nn][kq]) =                                                 \
        *(const uint4*)(SRC + (size_t)(n0 + nn) * DIM + (k0 + kq));           \
  }
#define LOAD_AFRAG(D, ARR)                                                    \
  D.q[0] = *(const uint4*)(&ARR[ml + lm][ah]);                                \
  D.q[1] = *(const uint4*)(&ARR[ml + lm][16 + ah]);
#define LOAD_BFRAG(D, ARR, ROW)                                               \
  D.q[0] = *(const uint4*)(&ARR[ROW][bh]);                                    \
  D.q[1] = *(const uint4*)(&ARR[ROW][bh + 16]);

  for (int k0 = 0; k0 < DIM; k0 += BK) {
    __syncthreads();
    FILL_A(vrh, Arh); FILL_A(vrl, Arl); FILL_A(vih, Aih); FILL_A(vil, Ail);
    FILL_B(srh, Brh); FILL_B(srl, Brl); FILL_B(sih, Bih); FILL_B(sil, Bil);
    __syncthreads();

    F16x16 arh, arl, aih, ail, nih, nil;
    LOAD_AFRAG(arh, Arh); LOAD_AFRAG(arl, Arl);
    LOAD_AFRAG(aih, Aih); LOAD_AFRAG(ail, Ail);
#pragma unroll
    for (int d = 0; d < 8; ++d) {                 // packed bf16 negate
      nih.u[d] = aih.u[d] ^ 0x80008000u;
      nil.u[d] = ail.u[d] ^ 0x80008000u;
    }
#pragma unroll
    for (int t = 0; t < 2; ++t) {
      int row = nwl + t * 16 + lm;
      F16x16 brh, brl, bih, bil;
      LOAD_BFRAG(brh, Brh, row); LOAD_BFRAG(brl, Brl, row);
      LOAD_BFRAG(bih, Bih, row); LOAD_BFRAG(bil, Bil, row);
      v8f ar = t ? accr1 : accr0;
      v8f ai = t ? acci1 : acci0;
      ar = wmma_bf16(arh.v, brh.v, ar);           // + Are*Bre (hh, hl, lh)
      ar = wmma_bf16(arh.v, brl.v, ar);
      ar = wmma_bf16(arl.v, brh.v, ar);
      ar = wmma_bf16(nih.v, bih.v, ar);           // - Aim*Bim
      ar = wmma_bf16(nih.v, bil.v, ar);
      ar = wmma_bf16(nil.v, bih.v, ar);
      ai = wmma_bf16(arh.v, bih.v, ai);           // + Are*Bim
      ai = wmma_bf16(arh.v, bil.v, ai);
      ai = wmma_bf16(arl.v, bih.v, ai);
      ai = wmma_bf16(aih.v, brh.v, ai);           // + Aim*Bre
      ai = wmma_bf16(aih.v, brl.v, ai);
      ai = wmma_bf16(ail.v, brh.v, ai);
      if (t) { accr1 = ar; acci1 = ai; } else { accr0 = ar; acci0 = ai; }
    }
  }

  // fused epilogue: |Y|^2, per-qubit signed reduction into shared ez
  const int row_base = m0 + ml + ((lane >> 4) << 3);   // + r gives global row
#pragma unroll
  for (int t = 0; t < 2; ++t) {
    const v8f& xre = t ? accr1 : accr0;
    const v8f& xim = t ? acci1 : acci0;
    float part[NQ];
#pragma unroll
    for (int q = 0; q < NQ; ++q) part[q] = 0.f;
#pragma unroll
    for (int r = 0; r < 8; ++r) {
      float re = xre[r], im = xim[r];
      float pw = re * re + im * im;
      int ig = row_base + r;
#pragma unroll
      for (int q = 0; q < NQ; ++q)
        part[q] = ((ig >> (11 - q)) & 1) ? (part[q] - pw) : (part[q] + pw);
    }
    int nl = nwl + t * 16 + lm;
#pragma unroll
    for (int q = 0; q < NQ; ++q) atomicAdd(&ez[nl][q], part[q]);
  }
  __syncthreads();
  for (int idx = tid; idx < BN * NQ; idx += 256) {
    int col = idx / NQ, q = idx - col * NQ;
    int j = n0 + col;                    // j = e*NB + b
    atomicAdd(&Eacc[(size_t)j * NQ + q], ez[col][q]);
  }
#undef FILL_A
#undef FILL_B
#undef LOAD_AFRAG
#undef LOAD_BFRAG
}

// ------------------------- small kernels -----------------------------------

__global__ void zero_kernel(float* __restrict__ p, int n) {
  int t = blockIdx.x * blockDim.x + threadIdx.x;
  if (t < n) p[t] = 0.f;
}

__global__ void finalize_kernel(const float* __restrict__ alpha,
                                const float* __restrict__ Eacc,
                                float* __restrict__ out) {
  int t = blockIdx.x * blockDim.x + threadIdx.x;
  if (t >= NB * NQ) return;
  float a0 = alpha[0], a1 = alpha[1], a2 = alpha[2], a3 = alpha[3];
  float mx = fmaxf(fmaxf(a0, a1), fmaxf(a2, a3));
  float w0 = expf(a0 - mx), w1 = expf(a1 - mx), w2 = expf(a2 - mx), w3 = expf(a3 - mx);
  float inv = 1.f / (w0 + w1 + w2 + w3);
  int b = t / NQ, q = t - b * NQ;
  float y = w0 * Eacc[((size_t)(0 * NB + b)) * NQ + q]
          + w1 * Eacc[((size_t)(1 * NB + b)) * NQ + q]
          + w2 * Eacc[((size_t)(2 * NB + b)) * NQ + q]
          + w3 * Eacc[((size_t)(3 * NB + b)) * NQ + q];
  out[t] = y * inv;
}

// ------------------------- fallback: direct LDS simulator ------------------

__global__ __launch_bounds__(256) void direct_sim_kernel(const float* __restrict__ features,
                                                         const float* __restrict__ theta,
                                                         const float* __restrict__ alpha,
                                                         float* __restrict__ out) {
  __shared__ float sre[DIM];
  __shared__ float sim[DIM];
  __shared__ float vr[NQ][2], vi[NQ][2];
  __shared__ float ezs[NQ];
  const int j = blockIdx.x;
  const int e = j >> 10;
  const int b = j & (NB - 1);
  const int tid = threadIdx.x;
  if (tid < NQ) {
    float v0r, v0i, v1r, v1i;
    enc_vec(e, features[b * NQ + tid], v0r, v0i, v1r, v1i);
    vr[tid][0] = v0r; vi[tid][0] = v0i;
    vr[tid][1] = v1r; vi[tid][1] = v1i;
    ezs[tid] = 0.f;
  }
  __syncthreads();
  for (int i = tid; i < DIM; i += 256) {
    float pr = 1.f, pi = 0.f;
#pragma unroll
    for (int q = 0; q < NQ; ++q) {
      int bit = (i >> (11 - q)) & 1;
      float wr_ = vr[q][bit], wi_ = vi[q][bit];
      float nr = pr * wr_ - pi * wi_;
      float ni = pr * wi_ + pi * wr_;
      pr = nr; pi = ni;
    }
    sre[i] = pr; sim[i] = pi;
  }
  apply_variational(sre, sim, theta, tid);
  __syncthreads();
  float part[NQ];
#pragma unroll
  for (int q = 0; q < NQ; ++q) part[q] = 0.f;
  for (int i = tid; i < DIM; i += 256) {
    float pw = sre[i] * sre[i] + sim[i] * sim[i];
#pragma unroll
    for (int q = 0; q < NQ; ++q)
      part[q] = ((i >> (11 - q)) & 1) ? (part[q] - pw) : (part[q] + pw);
  }
#pragma unroll
  for (int q = 0; q < NQ; ++q) atomicAdd(&ezs[q], part[q]);
  __syncthreads();
  if (tid < NQ) {
    float a0 = alpha[0], a1 = alpha[1], a2 = alpha[2], a3 = alpha[3];
    float mx = fmaxf(fmaxf(a0, a1), fmaxf(a2, a3));
    float w[NE] = {expf(a0 - mx), expf(a1 - mx), expf(a2 - mx), expf(a3 - mx)};
    float inv = 1.f / (w[0] + w[1] + w[2] + w[3]);
    atomicAdd(&out[b * NQ + tid], w[e] * inv * ezs[tid]);
  }
}

// ------------------------- launch ------------------------------------------

extern "C" void kernel_launch(void* const* d_in, const int* in_sizes, int n_in,
                              void* d_out, int out_size, void* d_ws, size_t ws_size,
                              hipStream_t stream) {
  (void)in_sizes; (void)n_in; (void)out_size;
  const float* features = (const float*)d_in[0];   // (1024, 12) f32
  const float* theta    = (const float*)d_in[1];   // (4, 12, 3) f32
  const float* alpha    = (const float*)d_in[2];   // (4,) f32
  float* out = (float*)d_out;                      // (1024, 12) f32

  const size_t MAT = (size_t)DIM * DIM;            // 16M elements per plane
  const size_t need = 8 * MAT * sizeof(unsigned short)
                    + (size_t)NE * NB * NQ * sizeof(float);

  if (ws_size >= need) {
    unsigned short* base = (unsigned short*)d_ws;
    unsigned short* vrh = base + 0 * MAT;
    unsigned short* vrl = base + 1 * MAT;
    unsigned short* vih = base + 2 * MAT;
    unsigned short* vil = base + 3 * MAT;
    unsigned short* srh = base + 4 * MAT;
    unsigned short* srl = base + 5 * MAT;
    unsigned short* sih = base + 6 * MAT;
    unsigned short* sil = base + 7 * MAT;
    float* Eacc = (float*)(base + 8 * MAT);

    build_v_kernel<<<DIM, 256, 0, stream>>>(theta, vrh, vrl, vih, vil);
    build_s_kernel<<<NCIRC, 256, 0, stream>>>(features, srh, srl, sih, sil);
    zero_kernel<<<(NE * NB * NQ + 255) / 256, 256, 0, stream>>>(Eacc, NE * NB * NQ);
    gemm_z_kernel<<<dim3(DIM / BM, NCIRC / BN), 256, 0, stream>>>(
        vrh, vrl, vih, vil, srh, srl, sih, sil, Eacc);
    finalize_kernel<<<(NB * NQ + 255) / 256, 256, 0, stream>>>(alpha, Eacc, out);
  } else {
    zero_kernel<<<(NB * NQ + 255) / 256, 256, 0, stream>>>(out, NB * NQ);
    direct_sim_kernel<<<NCIRC, 256, 0, stream>>>(features, theta, alpha, out);
  }
}
